// MPN_12910671692607
// MI455X (gfx1250) — matrix operand order repaired and checked by reading, not compiled, not run on hardware
//
#include <hip/hip_runtime.h>
#include <hip/hip_bf16.h>

typedef __attribute__((ext_vector_type(8)))  _Float16 v8h;
typedef __attribute__((ext_vector_type(16))) _Float16 v16h;
typedef __attribute__((ext_vector_type(8)))  float    v8f;
typedef __attribute__((ext_vector_type(4)))  unsigned int u32x4;
typedef __attribute__((ext_vector_type(8)))  int      i32x8;
typedef __attribute__((ext_vector_type(4)))  int      i32x4;

#define NN 100000
#define NE 1600000
#define NLAYER 20
#define ENC_NEG_INF 0x007FFFFFu
#define NTILES 19
#define WBLOB_DW (NTILES * 512 * 2 / 4)   // 4864 dwords = 19456 bytes

union V16u { v16h v; v8h h[2]; };

static __device__ __forceinline__ v16h cat16(v8h a, v8h b) {
  V16u u; u.h[0] = a; u.h[1] = b; return u.v;
}

// B fragment: element i of lane (h,e) = row[16*h + i]  (f16 row-major, 32 feats)
static __device__ __forceinline__ v16h frag_B(const _Float16* row, int h) {
  const v8h* p = (const v8h*)(row + 16 * h);
  return cat16(p[0], p[1]);
}

// A fragment from LDS-resident weight blob
static __device__ __forceinline__ v16h frag_A(const _Float16* Wlds, int tile, int lane) {
  const v8h* p = (const v8h*)(Wlds + tile * 512 + lane * 16);
  return cat16(p[0], p[1]);
}

static __device__ __forceinline__ v8f wmma16(v16h a, v16h b, v8f c) {
  return __builtin_amdgcn_wmma_f32_16x16x32_f16(false, a, false, b, (short)0, c,
                                                false, false);
}

static __device__ __forceinline__ unsigned encf(float f) {
  unsigned u = __float_as_uint(f);
  return u ^ ((u & 0x80000000u) ? 0xFFFFFFFFu : 0x80000000u);
}
static __device__ __forceinline__ float decf(unsigned u) {
  unsigned b = u ^ ((u & 0x80000000u) ? 0x80000000u : 0xFFFFFFFFu);
  return __uint_as_float(b);
}

// ---------------- weight packing: A-layout tiles of W^T ----------------
// A element i of lane (h,m): K = 16*h + i, M = 16*t + m
// tiles 0..8  : We1^T (96x48)   t=tile/3, c=tile%3
// tiles 9..12 : We2^T (48x32 pad 64) t=r/2, c=r%2
// tiles 13..16: Wv1^T (64x32)   t=r/2, c=r%2
// tiles 17..18: Wv2^T (32x32)   t=r, c=0
__global__ void pack_weights(const float* __restrict__ We1, const float* __restrict__ We2,
                             const float* __restrict__ Wv1, const float* __restrict__ Wv2,
                             _Float16* __restrict__ Wb) {
  int tile = blockIdx.x;
  int rem  = threadIdx.x;          // 0..511
  int lane = rem >> 4, i = rem & 15;
  int h = lane >> 4, m = lane & 15;
  int kk = 16 * h + i;
  float val = 0.f;
  if (tile < 9) {
    int t = tile / 3, c = tile % 3;
    int k = 32 * c + kk;                       // k < 96 always
    val = We1[k * 48 + 16 * t + m];
  } else if (tile < 13) {
    int r = tile - 9, t = r / 2, c = r % 2;
    int k = 32 * c + kk;
    if (k < 48) val = We2[k * 32 + 16 * t + m];
  } else if (tile < 17) {
    int r = tile - 13, t = r / 2, c = r % 2;
    int k = 32 * c + kk;                       // k < 64 always
    val = Wv1[k * 32 + 16 * t + m];
  } else {
    int t = tile - 17;
    val = Wv2[kk * 32 + 16 * t + m];
  }
  Wb[tile * 512 + lane * 16 + i] = (_Float16)val;
}

// ---------------- init ----------------
__global__ void init_nodes(const float* __restrict__ M, _Float16* __restrict__ Mh,
                           unsigned* __restrict__ Magg, long total) {
  long idx = (long)blockIdx.x * blockDim.x + threadIdx.x;
  if (idx < total) {
    Mh[idx] = (_Float16)M[idx];
    Magg[idx] = ENC_NEG_INF;
  }
}

__global__ void init_edges(const float* __restrict__ H, _Float16* __restrict__ Hbuf, long total) {
  long idx = (long)blockIdx.x * blockDim.x + threadIdx.x;
  if (idx < total) Hbuf[idx] = (_Float16)H[idx];
}

// ---------------- fused edge layer -------------------------------------
// 128 threads = 4 waves; each wave handles 2 subtiles of 16 edges (32 edges).
// Weight blob staged into LDS by the Tensor Data Mover (one DMA per block),
// tracked with TENSORcnt; each A tile feeds two WMMAs.
__global__ void __launch_bounds__(128) edge_layer(
    const _Float16* __restrict__ Mh, _Float16* __restrict__ Hbuf,
    unsigned* __restrict__ Magg,
    const int* __restrict__ src, const int* __restrict__ dst,
    const _Float16* __restrict__ Wb,
    const float* __restrict__ be1, const float* __restrict__ be2,
    const float* __restrict__ bv1, const float* __restrict__ bv2) {
  __shared__ _Float16 Wlds[NTILES * 512];   // 19456 B
  __shared__ _Float16 smem[4 * 4096];       // 4 waves * 2 subtiles * 2048 halfs

#if __has_builtin(__builtin_amdgcn_tensor_load_to_lds)
  // TDM: DMA the whole weight blob (4864 dwords, 2-D 1-row tile) into LDS.
  if (threadIdx.x < 32) {                    // wave 0 only (TDM ignores EXEC)
    unsigned long long ga = (unsigned long long)(const void*)Wb;
    unsigned ldsoff = (unsigned)(unsigned long long)(const void*)Wlds; // low 32b = LDS offset
    u32x4 g0;
    g0.x = 1u;                                             // count=1, user mode
    g0.y = ldsoff;                                         // D#.lds_addr
    g0.z = (unsigned)(ga & 0xFFFFFFFFull);                 // global_addr[31:0]
    g0.w = (unsigned)((ga >> 32) & 0x01FFFFFFull) | (2u << 30); // [56:32] | type=2
    i32x8 g1;
    g1[0] = 0x00020000;                 // workgroup_mask=0, data_size=4B
    g1[1] = (int)(WBLOB_DW << 16);      // tensor_dim0[15:0]  at bits [63:48]
    g1[2] = 0x00010000;                 // tensor_dim0 hi=0; tensor_dim1=1
    g1[3] = (int)(WBLOB_DW << 16);      // tile_dim0 at bits [127:112]
    g1[4] = 1;                          // tile_dim1=1, tile_dim2=0
    g1[5] = WBLOB_DW;                   // tensor_dim0_stride low 32
    g1[6] = 0;
    g1[7] = 0;
    i32x4 z4 = {0, 0, 0, 0};
#if __clang_major__ >= 23
    i32x8 z8 = {0, 0, 0, 0, 0, 0, 0, 0};
    __builtin_amdgcn_tensor_load_to_lds(g0, g1, z4, z4, z8, 0);
#else
    __builtin_amdgcn_tensor_load_to_lds(g0, g1, z4, z4, 0);
#endif
    __builtin_amdgcn_s_wait_tensorcnt(0);
  }
#else
  for (int o = threadIdx.x; o < NTILES * 512 / 8; o += 128)
    ((v8h*)Wlds)[o] = ((const v8h*)Wb)[o];
#endif
  __syncthreads();

  const int lane = threadIdx.x & 31;
  const int wib  = threadIdx.x >> 5;
  const int h    = lane >> 4;
  const int e    = lane & 15;
  const long base = ((long)blockIdx.x * 4 + wib) * 32;
  _Float16* SA = smem + wib * 4096;         // subtile 0 staging (2048 halfs)
  _Float16* SB = SA + 2048;                 // subtile 1 staging

  const long e0 = base + e, e1 = base + 16 + e;
  const int sn0 = src[e0], dn0 = dst[e0];
  const int sn1 = src[e1], dn1 = dst[e1];

  v16h bMi0 = frag_B(Mh + (long)dn0 * 32, h);
  v16h bMj0 = frag_B(Mh + (long)sn0 * 32, h);
  v16h bH0  = frag_B(Hbuf + e0 * 32, h);
  v16h bMi1 = frag_B(Mh + (long)dn1 * 32, h);
  v16h bMj1 = frag_B(Mh + (long)sn1 * 32, h);
  v16h bH1  = frag_B(Hbuf + e1 * 32, h);

  const v8f zacc = {0.f, 0.f, 0.f, 0.f, 0.f, 0.f, 0.f, 0.f};
  const v8h zh   = {(_Float16)0, (_Float16)0, (_Float16)0, (_Float16)0,
                    (_Float16)0, (_Float16)0, (_Float16)0, (_Float16)0};

  // ---- GEMM1: hidden1[48 x 16e] = We1^T @ [Mi;Mj;H], relu, stage [e][64]
#pragma unroll
  for (int t = 0; t < 3; ++t) {
    v16h a2 = frag_A(Wlds, t * 3 + 2, lane);
    v16h a1 = frag_A(Wlds, t * 3 + 1, lane);
    v16h a0 = frag_A(Wlds, t * 3 + 0, lane);
    v8f acc0 = wmma16(a0, bMi0, wmma16(a1, bMj0, wmma16(a2, bH0, zacc)));
    v8f acc1 = wmma16(a0, bMi1, wmma16(a1, bMj1, wmma16(a2, bH1, zacc)));
    const float* bp = be1 + 16 * t + 8 * h;
    v8h r0, r1;
#pragma unroll
    for (int j = 0; j < 8; ++j) {
      float v0 = acc0[j] + bp[j];
      float v1 = acc1[j] + bp[j];
      r0[j] = (_Float16)(v0 > 0.f ? v0 : 0.f);
      r1[j] = (_Float16)(v1 > 0.f ? v1 : 0.f);
    }
    *(v8h*)(SA + e * 64 + 16 * t + 8 * h) = r0;   // D rows 16t+8h+j contiguous
    *(v8h*)(SB + e * 64 + 16 * t + 8 * h) = r1;
  }
  *(v8h*)(SA + e * 64 + 48 + 8 * h) = zh;          // zero-pad feats 48..63
  *(v8h*)(SB + e * 64 + 48 + 8 * h) = zh;
  asm volatile("s_wait_dscnt 0x0" ::: "memory");

  v16h c0_0 = cat16(*(const v8h*)(SA + e * 64 + 0  + 16 * h),
                    *(const v8h*)(SA + e * 64 + 8  + 16 * h));
  v16h c1_0 = cat16(*(const v8h*)(SA + e * 64 + 32 + 16 * h),
                    *(const v8h*)(SA + e * 64 + 40 + 16 * h));
  v16h c0_1 = cat16(*(const v8h*)(SB + e * 64 + 0  + 16 * h),
                    *(const v8h*)(SB + e * 64 + 8  + 16 * h));
  v16h c1_1 = cat16(*(const v8h*)(SB + e * 64 + 32 + 16 * h),
                    *(const v8h*)(SB + e * 64 + 40 + 16 * h));

  // ---- GEMM2: H_new[32 x 16e] = We2^T @ hidden1, +bias, stage [e][32]
#pragma unroll
  for (int t = 0; t < 2; ++t) {
    v16h ahi = frag_A(Wlds, 9 + t * 2 + 1, lane);
    v16h alo = frag_A(Wlds, 9 + t * 2 + 0, lane);
    v8f acc0 = wmma16(alo, c0_0, wmma16(ahi, c1_0, zacc));
    v8f acc1 = wmma16(alo, c0_1, wmma16(ahi, c1_1, zacc));
    const float* bp = be2 + 16 * t + 8 * h;
    v8h r0, r1;
#pragma unroll
    for (int j = 0; j < 8; ++j) {
      r0[j] = (_Float16)(acc0[j] + bp[j]);
      r1[j] = (_Float16)(acc1[j] + bp[j]);
    }
    *(v8h*)(SA + 1024 + e * 32 + 16 * t + 8 * h) = r0;
    *(v8h*)(SB + 1024 + e * 32 + 16 * t + 8 * h) = r1;
  }
  asm volatile("s_wait_dscnt 0x0" ::: "memory");

  v8h hnA0 = *(const v8h*)(SA + 1024 + e * 32 + 0 + 16 * h);
  v8h hnA1 = *(const v8h*)(SA + 1024 + e * 32 + 8 + 16 * h);
  v8h hnB0 = *(const v8h*)(SB + 1024 + e * 32 + 0 + 16 * h);
  v8h hnB1 = *(const v8h*)(SB + 1024 + e * 32 + 8 + 16 * h);
  v16h bHn0 = cat16(hnA0, hnA1);
  v16h bHn1 = cat16(hnB0, hnB1);

  // write H_new back (in place; this wave owns these rows)
  *(v8h*)(Hbuf + e0 * 32 + 16 * h + 0) = hnA0;
  *(v8h*)(Hbuf + e0 * 32 + 16 * h + 8) = hnA1;
  *(v8h*)(Hbuf + e1 * 32 + 16 * h + 0) = hnB0;
  *(v8h*)(Hbuf + e1 * 32 + 16 * h + 8) = hnB1;

  // ---- GEMM3: hid_v[32 x 16e] = Wv1^T @ [Mi; H_new], relu, stage [e][32]
#pragma unroll
  for (int t = 0; t < 2; ++t) {
    v16h ahi = frag_A(Wlds, 13 + t * 2 + 1, lane);
    v16h alo = frag_A(Wlds, 13 + t * 2 + 0, lane);
    v8f acc0 = wmma16(alo, bMi0, wmma16(ahi, bHn0, zacc));
    v8f acc1 = wmma16(alo, bMi1, wmma16(ahi, bHn1, zacc));
    const float* bp = bv1 + 16 * t + 8 * h;
    v8h r0, r1;
#pragma unroll
    for (int j = 0; j < 8; ++j) {
      float v0 = acc0[j] + bp[j];
      float v1 = acc1[j] + bp[j];
      r0[j] = (_Float16)(v0 > 0.f ? v0 : 0.f);
      r1[j] = (_Float16)(v1 > 0.f ? v1 : 0.f);
    }
    *(v8h*)(SA + 1536 + e * 32 + 16 * t + 8 * h) = r0;
    *(v8h*)(SB + 1536 + e * 32 + 16 * t + 8 * h) = r1;
  }
  asm volatile("s_wait_dscnt 0x0" ::: "memory");

  v16h c3_0 = cat16(*(const v8h*)(SA + 1536 + e * 32 + 0 + 16 * h),
                    *(const v8h*)(SA + 1536 + e * 32 + 8 + 16 * h));
  v16h c3_1 = cat16(*(const v8h*)(SB + 1536 + e * 32 + 0 + 16 * h),
                    *(const v8h*)(SB + 1536 + e * 32 + 8 + 16 * h));

  // ---- GEMM4: M_msg[32 x 16e] = Wv2^T @ hid_v, +bias, scatter-max to dst
#pragma unroll
  for (int t = 0; t < 2; ++t) {
    v16h at = frag_A(Wlds, 17 + t, lane);
    v8f acc0 = wmma16(at, c3_0, zacc);
    v8f acc1 = wmma16(at, c3_1, zacc);
    const float* bp = bv2 + 16 * t + 8 * h;
#pragma unroll
    for (int j = 0; j < 8; ++j) {
      float v0 = acc0[j] + bp[j];
      float v1 = acc1[j] + bp[j];
      atomicMax(&Magg[(long)dn0 * 32 + 16 * t + 8 * h + j], encf(v0));
      atomicMax(&Magg[(long)dn1 * 32 + 16 * t + 8 * h + j], encf(v1));
    }
  }
}

// ---------------- node update: decode max, empty->0, refresh Mh, reset agg --
__global__ void node_update(unsigned* __restrict__ Magg, _Float16* __restrict__ Mh, long total) {
  long idx = (long)blockIdx.x * blockDim.x + threadIdx.x;
  if (idx < total) {
    unsigned u = Magg[idx];
    float v = (u == ENC_NEG_INF) ? 0.f : decf(u);
    Mh[idx] = (_Float16)v;
    Magg[idx] = ENC_NEG_INF;
  }
}

// ---------------- final: relu(H @ Wo + bo) ----------------
__global__ void out_kernel(const _Float16* __restrict__ Hbuf, const float* __restrict__ Wo,
                           const float* __restrict__ bo, float* __restrict__ out, long E) {
  long e = (long)blockIdx.x * blockDim.x + threadIdx.x;
  if (e < E) {
    float s = bo[0];
#pragma unroll
    for (int f = 0; f < 32; ++f) s += (float)Hbuf[e * 32 + f] * Wo[f];
    out[e] = s > 0.f ? s : 0.f;
  }
}

extern "C" void kernel_launch(void* const* d_in, const int* in_sizes, int n_in,
                              void* d_out, int out_size, void* d_ws, size_t ws_size,
                              hipStream_t stream) {
  const float* M   = (const float*)d_in[0];
  const float* H   = (const float*)d_in[1];
  const int*   ei  = (const int*)d_in[2];
  const float* We1 = (const float*)d_in[3];
  const float* be1 = (const float*)d_in[4];
  const float* We2 = (const float*)d_in[5];
  const float* be2 = (const float*)d_in[6];
  const float* Wv1 = (const float*)d_in[7];
  const float* bv1 = (const float*)d_in[8];
  const float* Wv2 = (const float*)d_in[9];
  const float* bv2 = (const float*)d_in[10];
  const float* Wo  = (const float*)d_in[11];
  const float* bo  = (const float*)d_in[12];

  char* ws = (char*)d_ws;
  _Float16* Wb   = (_Float16*)ws;                                   // 19456 B (reserve 32 KB)
  _Float16* Mh   = (_Float16*)(ws + 32768);                         // 6.4 MB
  unsigned* Magg = (unsigned*)(ws + 32768 + 6400000);               // 12.8 MB
  _Float16* Hbuf = (_Float16*)(ws + 32768 + 6400000 + 12800000);    // 102.4 MB

  const int* src = ei;            // edge_index[0]
  const int* dst = ei + NE;       // edge_index[1]

  pack_weights<<<19, 512, 0, stream>>>(We1, We2, Wv1, Wv2, Wb);

  long nTot = (long)NN * 32;
  long eTot = (long)NE * 32;
  init_nodes<<<(nTot + 255) / 256, 256, 0, stream>>>(M, Mh, Magg, nTot);
  init_edges<<<(eTot + 255) / 256, 256, 0, stream>>>(H, Hbuf, eTot);

  for (int l = 0; l < NLAYER; ++l) {
    // 4 waves/block, 32 edges/wave -> 128 edges per block
    edge_layer<<<NE / 128, 128, 0, stream>>>(Mh, Hbuf, Magg, src, dst, Wb,
                                             be1, be2, bv1, bv2);
    node_update<<<(nTot + 255) / 256, 256, 0, stream>>>(Magg, Mh, nTot);
  }

  out_kernel<<<(NE + 255) / 256, 256, 0, stream>>>(Hbuf, Wo, bo, (float*)d_out, NE);
}
